// RnnEncoder_13048110646093
// MI455X (gfx1250) — matrix operand order repaired and checked by reading, not compiled, not run on hardware
//
#include <hip/hip_runtime.h>
#include <hip/hip_bf16.h>

#define Hdim  744
#define HP    768          // H padded to multiple of 32
#define Tlen  256
#define Bsz   256
#define Cclip 16
#define Wrow  800
#define TOUT  224          // T - 2C
#define OUTW  1488         // 2H
#define NT    48           // HP / 16  (N tiles per gate)
#define KT    24           // HP / 32  (K steps total)
#define KTH   12           // K steps per wave half
#define MT    16           // B / 16   (M tiles)
#define NFRAG (2*2*3*NT*KT)        // dirs * {Wih,Whh} * gates * NT * KT = 13824
#define FRAG_ELEMS 512             // 32 lanes * 16 bf16
#define FRAG_BYTES ((size_t)NFRAG * FRAG_ELEMS * 2)
#define HSTATE (2 * Bsz * HP)                 // floats per ping buffer (both dirs)
#define HSF_BYTES  ((size_t)2 * HSTATE * 4)   // 2 pings of fp32 h
#define HBF_BYTES  ((size_t)2 * HSTATE * 2)   // 2 pings of bf16 h
#define XBF_ELEMS  ((size_t)Bsz * Tlen * HP)
#define XBF_BYTES  (XBF_ELEMS * 2)

typedef __attribute__((ext_vector_type(16))) __bf16 v16bf;
typedef __attribute__((ext_vector_type(8)))  __bf16 v8bf;
typedef __attribute__((ext_vector_type(8)))  float  v8f;

// ---------------------------------------------------------------------------
// Stage weights into WMMA B-fragment order, bf16, zero-padded 744 -> 768.
// B layout (16-bit, 32x16): lane l holds column n = l%16;
// element e holds K = kt*32 + (l<16 ? 0 : 8) + (e&7) + (e<8 ? 0 : 16).
// B(k,n) must equal W[g*744 + n][k]  (gemm is  act @ W^T).
// ---------------------------------------------------------------------------
__global__ __launch_bounds__(256) void stage_w(__bf16* __restrict__ frag,
    const float* __restrict__ Wih_f, const float* __restrict__ Whh_f,
    const float* __restrict__ Wih_b, const float* __restrict__ Whh_b) {
  size_t idx = (size_t)blockIdx.x * 256 + threadIdx.x;
  if (idx >= (size_t)NFRAG * FRAG_ELEMS) return;
  int e    = (int)(idx & 15);
  int lane = (int)((idx >> 4) & 31);
  int f    = (int)(idx >> 9);
  int kt  = f % KT; f /= KT;
  int nt  = f % NT; f /= NT;
  int g   = f % 3;  f /= 3;
  int mat = f & 1;
  int dir = f >> 1;
  const float* W = dir ? (mat ? Whh_b : Wih_b) : (mat ? Whh_f : Wih_f);
  int n = nt * 16 + (lane & 15);
  int k = kt * 32 + ((lane < 16) ? 0 : 8) + (e & 7) + ((e < 8) ? 0 : 16);
  float v = 0.f;
  if (n < Hdim && k < Hdim) v = W[(size_t)(g * Hdim + n) * Hdim + k];
  frag[idx] = (__bf16)v;
}

// Stage x: windows[b][t][0:744] -> bf16 xbf[b][t][0:768], zero padded.
__global__ __launch_bounds__(256) void stage_x(__bf16* __restrict__ xbf,
                                               const float* __restrict__ windows) {
  size_t idx = (size_t)blockIdx.x * 256 + threadIdx.x;
  if (idx >= XBF_ELEMS) return;
  int k = (int)(idx % HP);
  size_t bt = idx / HP;
  float v = (k < Hdim) ? windows[bt * Wrow + k] : 0.f;
  xbf[idx] = (__bf16)v;
}

__global__ __launch_bounds__(256) void zero_f32(float* __restrict__ p, size_t n) {
  size_t i = (size_t)blockIdx.x * 256 + threadIdx.x;
  if (i < n) p[i] = 0.f;
}

__device__ __forceinline__ float sigm(float x) {
  return 1.f / (1.f + __expf(-x));
}

// A layout (16-bit, 16x32): lane l holds row m = l%16;
// elements 0..7 -> K = kbase + (l<16?0:8) + e; elements 8..15 -> +16.
__device__ __forceinline__ v16bf load_a_bf(const __bf16* __restrict__ row,
                                           int kbase, int koff) {
  v8bf lo = *(const v8bf*)(row + kbase + koff);
  v8bf hi = *(const v8bf*)(row + kbase + koff + 16);
  v16bf a;
#pragma unroll
  for (int e = 0; e < 8; ++e) { a[e] = lo[e]; a[e + 8] = hi[e]; }
  return a;
}

// Fallback path: convert fp32 on the fly, masked above kmax (kmax relative to row).
__device__ __forceinline__ v16bf load_a_f32(const float* __restrict__ row,
                                            int kbase, int koff, int kmax) {
  v16bf a;
#pragma unroll
  for (int e = 0; e < 8; ++e) {
    int k0 = kbase + koff + e;
    int k1 = k0 + 16;
    a[e]     = (k0 < kmax) ? (__bf16)row[k0] : (__bf16)0.f;
    a[e + 8] = (k1 < kmax) ? (__bf16)row[k1] : (__bf16)0.f;
  }
  return a;
}

// ---------------------------------------------------------------------------
// One GRU time step, both directions fused. Each 16x16 (batch x hid) tile is
// computed by a PAIR of waves splitting K (half 0: kt 0-11, half 1: kt 12-23)
// to double occupancy and halve the per-wave L2-latency chain. Partial sums
// combine through LDS; half 0 runs the gating epilogue.
// ---------------------------------------------------------------------------
template <bool XBF>
__global__ __launch_bounds__(256) void gru_step(
    const float*  __restrict__ windows,
    const __bf16* __restrict__ xbf,
    const __bf16* __restrict__ frag,
    const float*  __restrict__ hprev,
    const __bf16* __restrict__ hbf_prev,
    float*        __restrict__ hnext,
    __bf16*       __restrict__ hbf_next,
    const float*  __restrict__ bih_f, const float* __restrict__ bhh_f,
    const float*  __restrict__ bih_b, const float* __restrict__ bhh_b,
    float*        __restrict__ out, int t) {
  __shared__ __align__(32) float lds_acc[4][6][32][8];   // 24 KB / block

  const int lane = threadIdx.x & 31;
  const int pIn  = threadIdx.x >> 6;                     // pair within block 0..3
  const int half = (threadIdx.x >> 5) & 1;               // K half
  const int pid  = blockIdx.x * 4 + pIn;                 // 0 .. 1535
  const int dir  = pid / (MT * NT);
  const int rem  = pid % (MT * NT);
  const int mt   = rem / NT;
  const int nt   = rem % NT;
  const int t_x  = dir ? (Tlen - 1 - t) : t;

  const int m    = lane & 15;
  const int koff = (lane < 16) ? 0 : 8;
  const int bm   = mt * 16 + m;                          // batch row for A frag
  const int kofs = half * (KTH * 32);                    // element offset of this K half
  const float*  xrow_f = windows  + ((size_t)bm * Tlen + t_x) * Wrow + kofs;
  const __bf16* xrow_b = xbf      + ((size_t)bm * Tlen + t_x) * HP   + kofs;
  const __bf16* hrow   = hbf_prev + (size_t)(dir * Bsz + bm) * HP    + kofs;
  const int kmax_loc = Hdim - kofs;                      // for fallback masking

  v8f ai0 = {}, ai1 = {}, ai2 = {};   // gi: r, z, n
  v8f ah0 = {}, ah1 = {}, ah2 = {};   // gh: r, z, n

  const size_t gstep  = (size_t)NT * KT * FRAG_ELEMS;    // stride between gates
  const size_t lane16 = (size_t)lane * 16;
  const size_t hofs   = (size_t)half * KTH * FRAG_ELEMS;
  const size_t base_i0 = ((size_t)(((dir * 2 + 0) * 3 + 0) * NT + nt) * KT) * FRAG_ELEMS + hofs + lane16;
  const size_t base_h0 = ((size_t)(((dir * 2 + 1) * 3 + 0) * NT + nt) * KT) * FRAG_ELEMS + hofs + lane16;

#define LOADBI(kk, gg) (*(const v16bf*)(frag + base_i0 + (size_t)(gg) * gstep + (size_t)(kk) * FRAG_ELEMS))
#define LOADBH(kk, gg) (*(const v16bf*)(frag + base_h0 + (size_t)(gg) * gstep + (size_t)(kk) * FRAG_ELEMS))
#define LOADAX(kk) (XBF ? load_a_bf(xrow_b, (kk) * 32, koff) \
                        : load_a_f32(xrow_f, (kk) * 32, koff, kmax_loc))

  // Prologue: fragments for local kt = 0.
  v16bf ax_c = LOADAX(0);
  v16bf ah_c = load_a_bf(hrow, 0, koff);
  v16bf bi_c[3], bh_c[3];
#pragma unroll
  for (int g = 0; g < 3; ++g) { bi_c[g] = LOADBI(0, g); bh_c[g] = LOADBH(0, g); }

#pragma unroll
  for (int kt = 0; kt < KTH; ++kt) {
    const int kn = (kt + 1 < KTH) ? (kt + 1) : 0;  // last iter prefetch is dummy
    // Issue next iteration's loads first so current WMMAs overlap them.
    v16bf ax_n = LOADAX(kn);
    v16bf ah_n = load_a_bf(hrow, kn * 32, koff);
    v16bf bi_n[3], bh_n[3];
#pragma unroll
    for (int g = 0; g < 3; ++g) { bi_n[g] = LOADBI(kn, g); bh_n[g] = LOADBH(kn, g); }

    ai0 = __builtin_amdgcn_wmma_f32_16x16x32_bf16(false, ax_c, false, bi_c[0], (short)0, ai0, false, false);
    ai1 = __builtin_amdgcn_wmma_f32_16x16x32_bf16(false, ax_c, false, bi_c[1], (short)0, ai1, false, false);
    ai2 = __builtin_amdgcn_wmma_f32_16x16x32_bf16(false, ax_c, false, bi_c[2], (short)0, ai2, false, false);
    ah0 = __builtin_amdgcn_wmma_f32_16x16x32_bf16(false, ah_c, false, bh_c[0], (short)0, ah0, false, false);
    ah1 = __builtin_amdgcn_wmma_f32_16x16x32_bf16(false, ah_c, false, bh_c[1], (short)0, ah1, false, false);
    ah2 = __builtin_amdgcn_wmma_f32_16x16x32_bf16(false, ah_c, false, bh_c[2], (short)0, ah2, false, false);

    ax_c = ax_n; ah_c = ah_n;
#pragma unroll
    for (int g = 0; g < 3; ++g) { bi_c[g] = bi_n[g]; bh_c[g] = bh_n[g]; }
  }
#undef LOADBI
#undef LOADBH
#undef LOADAX

  // Combine the two K halves through LDS.
  if (half == 1) {
    *(v8f*)&lds_acc[pIn][0][lane][0] = ai0;
    *(v8f*)&lds_acc[pIn][1][lane][0] = ai1;
    *(v8f*)&lds_acc[pIn][2][lane][0] = ai2;
    *(v8f*)&lds_acc[pIn][3][lane][0] = ah0;
    *(v8f*)&lds_acc[pIn][4][lane][0] = ah1;
    *(v8f*)&lds_acc[pIn][5][lane][0] = ah2;
  }
  __syncthreads();
  if (half == 1) return;

  ai0 += *(const v8f*)&lds_acc[pIn][0][lane][0];
  ai1 += *(const v8f*)&lds_acc[pIn][1][lane][0];
  ai2 += *(const v8f*)&lds_acc[pIn][2][lane][0];
  ah0 += *(const v8f*)&lds_acc[pIn][3][lane][0];
  ah1 += *(const v8f*)&lds_acc[pIn][4][lane][0];
  ah2 += *(const v8f*)&lds_acc[pIn][5][lane][0];

  // C/D layout: VGPR v, lane l -> M = v + (l<16 ? 0 : 8), N = l % 16.
  const int jj = nt * 16 + (lane & 15);
  const bool jv = jj < Hdim;
  const float* bih = dir ? bih_b : bih_f;
  const float* bhh = dir ? bhh_b : bhh_f;
  const float bir = jv ? bih[jj]            : 0.f;
  const float biz = jv ? bih[Hdim + jj]     : 0.f;
  const float bin = jv ? bih[2 * Hdim + jj] : 0.f;
  const float bhr = jv ? bhh[jj]            : 0.f;
  const float bhz = jv ? bhh[Hdim + jj]     : 0.f;
  const float bhn = jv ? bhh[2 * Hdim + jj] : 0.f;
  const bool do_out = (t >= Cclip) && (t < Tlen - Cclip);
  const int mo = (lane < 16) ? 0 : 8;

#pragma unroll
  for (int v = 0; v < 8; ++v) {
    int b = mt * 16 + v + mo;
    size_t hidx = (size_t)(dir * Bsz + b) * HP + jj;
    float hp = hprev[hidx];
    float r  = sigm(ai0[v] + bir + ah0[v] + bhr);
    float z  = sigm(ai1[v] + biz + ah1[v] + bhz);
    float nn = tanhf(ai2[v] + bin + r * (ah2[v] + bhn));
    float hn = (1.f - z) * nn + z * hp;
    hnext[hidx] = hn;
    hbf_next[hidx] = (__bf16)hn;
    if (do_out && jv) {
      float xv = windows[((size_t)b * Tlen + t_x) * Wrow + jj];
      out[((size_t)b * TOUT + (t - Cclip)) * OUTW + dir * Hdim + jj] = hn + xv;
    }
  }
}

extern "C" void kernel_launch(void* const* d_in, const int* in_sizes, int n_in,
                              void* d_out, int out_size, void* d_ws, size_t ws_size,
                              hipStream_t stream) {
  (void)in_sizes; (void)n_in; (void)out_size;
  const float* windows = (const float*)d_in[0];
  const float* Wih_f   = (const float*)d_in[1];
  const float* Whh_f   = (const float*)d_in[2];
  const float* bih_f   = (const float*)d_in[3];
  const float* bhh_f   = (const float*)d_in[4];
  const float* Wih_b   = (const float*)d_in[5];
  const float* Whh_b   = (const float*)d_in[6];
  const float* bih_b   = (const float*)d_in[7];
  const float* bhh_b   = (const float*)d_in[8];
  float* out = (float*)d_out;

  const size_t need_base = FRAG_BYTES + HSF_BYTES + HBF_BYTES;
  const size_t need_full = need_base + XBF_BYTES;
  if (ws_size < need_base) return;
  const bool use_x = (ws_size >= need_full);

  __bf16* frag = (__bf16*)d_ws;
  float*  hbuf = (float*)((char*)d_ws + FRAG_BYTES);
  __bf16* hbf  = (__bf16*)((char*)d_ws + FRAG_BYTES + HSF_BYTES);
  __bf16* xbf  = (__bf16*)((char*)d_ws + need_base);

  {
    size_t total = (size_t)NFRAG * FRAG_ELEMS;
    stage_w<<<(int)((total + 255) / 256), 256, 0, stream>>>(
        frag, Wih_f, Whh_f, Wih_b, Whh_b);
  }
  {
    // Zero both fp32 h pings and both bf16 h pings (contiguous region).
    size_t nf = (HSF_BYTES + HBF_BYTES) / 4;
    zero_f32<<<(int)((nf + 255) / 256), 256, 0, stream>>>(hbuf, nf);
  }
  if (use_x) {
    stage_x<<<(int)((XBF_ELEMS + 255) / 256), 256, 0, stream>>>(xbf, windows);
  }

  const int nblocks = (2 * MT * NT) / 4;   // 1536 tile-pairs / 4 pairs per block
  for (int t = 0; t < Tlen; ++t) {
    const float*  hpf = hbuf + (size_t)(t & 1) * HSTATE;
    float*        hnf = hbuf + (size_t)((t ^ 1) & 1) * HSTATE;
    const __bf16* hpb = hbf + (size_t)(t & 1) * HSTATE;
    __bf16*       hnb = hbf + (size_t)((t ^ 1) & 1) * HSTATE;
    if (use_x) {
      gru_step<true><<<nblocks, 256, 0, stream>>>(
          windows, xbf, frag, hpf, hpb, hnf, hnb,
          bih_f, bhh_f, bih_b, bhh_b, out, t);
    } else {
      gru_step<false><<<nblocks, 256, 0, stream>>>(
          windows, xbf, frag, hpf, hpb, hnf, hnb,
          bih_f, bhh_f, bih_b, bhh_b, out, t);
    }
  }
}
